// CausalSelfAttention_5634997092961
// MI455X (gfx1250) — matrix operand order repaired and checked by reading, not compiled, not run on hardware
//
#include <hip/hip_runtime.h>

// Problem constants (from reference)
#define BATCH 4
#define SEQ_T 2048
#define EMB   1024
#define NHEAD 16
#define HDIM  64
#define MTOT  (BATCH*SEQ_T)   // 8192

typedef __attribute__((ext_vector_type(16))) __bf16 bf16x16;
typedef __attribute__((ext_vector_type(8)))  float  f32x8;
typedef int v4i __attribute__((vector_size(16)));   // async-lds builtin param type
typedef unsigned int u32x4 __attribute__((vector_size(16)));
typedef int i32x8 __attribute__((vector_size(32)));
typedef int i32x4 __attribute__((vector_size(16)));

union Frag { unsigned int u[8]; bf16x16 v; };

static __device__ __forceinline__ unsigned short f2bf(float f) {
  unsigned int u = __builtin_bit_cast(unsigned int, f);
  u += 0x7FFFu + ((u >> 16) & 1u);          // round-to-nearest-even
  return (unsigned short)(u >> 16);
}
static __device__ __forceinline__ float bf2f(unsigned short h) {
  unsigned int u = ((unsigned int)h) << 16;
  return __builtin_bit_cast(float, u);
}
static __device__ __forceinline__ unsigned int pack2bf(float a, float b) {
  return (unsigned int)f2bf(a) | ((unsigned int)f2bf(b) << 16);
}

// Async global->LDS b128 copy (ASYNCcnt-tracked) with sync fallback.
#if __has_builtin(__builtin_amdgcn_global_load_async_to_lds_b128) && \
    __has_builtin(__builtin_amdgcn_s_wait_asynccnt)
static __device__ __forceinline__ void async_copy16(void* lds, const void* g) {
  __builtin_amdgcn_global_load_async_to_lds_b128(
      (__attribute__((address_space(1))) v4i*)(v4i*)(void*)(uintptr_t)g,
      (__attribute__((address_space(3))) v4i*)(v4i*)lds, 0, 0);
}
static __device__ __forceinline__ void async_wait() {
  __builtin_amdgcn_s_wait_asynccnt(0);
}
#else
static __device__ __forceinline__ void async_copy16(void* lds, const void* g) {
  *(uint4*)lds = *(const uint4*)g;
}
static __device__ __forceinline__ void async_wait() {}
#endif

#define LDK 72   // bf16 elems; 144B row stride, 16B aligned

// TDM: DMA a 64x64 bf16 tile (row stride 64 elems) into LDS with LDK padding.
// D# layout per CDNA5 ISA. Pad: +4 dwords every 32 dwords.
// This toolchain exposes the 6-arg builtin: (g0, g1, g2, g3, g4, cpol).
#if __has_builtin(__builtin_amdgcn_tensor_load_to_lds) && \
    __has_builtin(__builtin_amdgcn_s_wait_tensorcnt)
#define HAVE_TDM 1
static __device__ __forceinline__ void tdm_load_tile64(void* lds, const void* g) {
  unsigned int lds_addr = (unsigned int)(uintptr_t)
      (__attribute__((address_space(3))) void*)lds;
  unsigned long long ga = (unsigned long long)(uintptr_t)g;
  u32x4 g0 = { 1u,                                  // count=1, user mode
               lds_addr,
               (unsigned int)ga,                    // global_addr[31:0]
               ((unsigned int)(ga >> 32) & 0x01FFFFFFu) | 0x80000000u }; // type=2
  i32x8 g1 = { (int)((1u << 16)      // data_size = 2 bytes
                   | (1u << 20)      // pad_enable
                   | (4u << 22)      // pad_interval: 32 dwords (one 64-elem row)
                   | (3u << 25)),    // pad_amount: 4 dwords (8 bf16)
               64 << 16,             // tensor_dim0 = 64
               64 << 16,             // tensor_dim1 = 64
               64 << 16,             // tile_dim0   = 64
               64,                   // tile_dim1   = 64, tile_dim2 = 0
               64,                   // tensor_dim0_stride = 64
               0, 0 };               // tensor_dim1_stride = 0
  i32x4 gz  = { 0, 0, 0, 0 };
  i32x8 gz8 = { 0, 0, 0, 0, 0, 0, 0, 0 };
  __builtin_amdgcn_tensor_load_to_lds(g0, g1, gz, gz, gz8, 0);
}
static __device__ __forceinline__ void tdm_wait() {
  __builtin_amdgcn_s_wait_tensorcnt(0);
}
#else
#define HAVE_TDM 0
#endif

// ---------------------------------------------------------------------------
// One-shot f32 -> bf16 conversion of x (done once; GEMMs then DMA bf16 tiles)
// ---------------------------------------------------------------------------
__global__ __launch_bounds__(256)
void cvt_f32_bf16(const float* __restrict__ x, unsigned short* __restrict__ o) {
  size_t i = ((size_t)blockIdx.x * 256 + threadIdx.x) * 4;
  float4 f = *(const float4*)(x + i);
  uint2 p;
  p.x = pack2bf(f.x, f.y);
  p.y = pack2bf(f.z, f.w);
  *(uint2*)(o + i) = p;
}

// ---------------------------------------------------------------------------
// GEMM: C[M,N] = A[M,K] * W[K,N] + bias, K=N=1024, M=8192, A bf16.
// mode 0: plain row-major bf16 store (output projection)
// mode 1: scatter to [B,H,T,D] with RoPE (Q,K)
// mode 2: scatter to [B,H,T,D] without RoPE (V)
// ---------------------------------------------------------------------------
#define BM 128
#define BN 64
#define BK 32
#define LDA 40   // bf16 elems; 80B row stride, 16B aligned
#define LDB 40

__global__ __launch_bounds__(256, 2)
void gemm_bf16_wmma(const unsigned short* __restrict__ A,
                    const unsigned short* __restrict__ W,
                    const unsigned short* __restrict__ bias,
                    unsigned short* __restrict__ dst, int mode)
{
  __shared__ unsigned short As[BM * LDA];        // [m][k]
  __shared__ unsigned short Bs[BN * LDB];        // [n][k] (W transposed)

  const int tid  = threadIdx.x;
  const int lane = tid & 31, wid = tid >> 5;
  const int half = lane >> 4, idx = lane & 15;
  const int waveM = (wid & 3) * 32;
  const int waveN = (wid >> 2) * 32;
  const int m0 = blockIdx.y * BM;
  const int n0 = blockIdx.x * BN;

  f32x8 acc[2][2];
  #pragma unroll
  for (int a = 0; a < 2; ++a)
    #pragma unroll
    for (int b = 0; b < 2; ++b)
      #pragma unroll
      for (int i = 0; i < 8; ++i) acc[a][b][i] = 0.f;

  for (int k0 = 0; k0 < EMB; k0 += BK) {
    // ---- stage A tile via async DMA ----
    {
      int r = tid >> 2, c = (tid & 3) * 8;
      #pragma unroll
      for (int i = 0; i < 2; ++i) {
        int row = r + 64 * i;
        async_copy16(&As[row * LDA + c],
                     A + (size_t)(m0 + row) * EMB + k0 + c);
      }
    }
    // ---- stage B tile transposed: Bs[n][k] = W[k0+k][n0+n] ----
    {
      int kr = tid >> 3, nc = (tid & 7) * 8;
      unsigned short tmp[8];
      *(uint4*)tmp = *(const uint4*)(W + (size_t)(k0 + kr) * EMB + n0 + nc);
      #pragma unroll
      for (int j = 0; j < 8; ++j) Bs[(nc + j) * LDB + kr] = tmp[j];
    }
    async_wait();
    __syncthreads();

    // ---- gather fragments & WMMA ----
    Frag a[2], b[2];
    #pragma unroll
    for (int tm = 0; tm < 2; ++tm) {
      const unsigned short* p = &As[(waveM + tm * 16 + idx) * LDA];
      #pragma unroll
      for (int j = 0; j < 8; ++j) {
        int ko = (half << 4) + (j << 1);
        a[tm].u[j] = *(const unsigned int*)(p + ko);
      }
    }
    #pragma unroll
    for (int tn = 0; tn < 2; ++tn) {
      const unsigned short* p = &Bs[(waveN + tn * 16 + idx) * LDB];
      #pragma unroll
      for (int j = 0; j < 8; ++j)
        b[tn].u[j] = *(const unsigned int*)(p + (half << 4) + (j << 1));
    }
    #pragma unroll
    for (int tm = 0; tm < 2; ++tm)
      #pragma unroll
      for (int tn = 0; tn < 2; ++tn)
        acc[tm][tn] = __builtin_amdgcn_wmma_f32_16x16x32_bf16(
            false, a[tm].v, false, b[tn].v, (short)0, acc[tm][tn], false, false);
    __syncthreads();
  }

  // ---- epilogue: bias (+ RoPE) + scatter ----
  #pragma unroll
  for (int tn = 0; tn < 2; ++tn) {
    const int col = n0 + waveN + tn * 16 + idx;
    const float bv = bf2f(bias[col]);
    const int d = col & (HDIM - 1);
    const int h = col >> 6;
    float inv_freq = 0.f;
    if (mode == 1) inv_freq = __powf(10000.f, -(float)(d >> 1) / 32.f);
    #pragma unroll
    for (int tm = 0; tm < 2; ++tm) {
      #pragma unroll
      for (int r = 0; r < 8; ++r) {
        int row = m0 + waveM + tm * 16 + half * 8 + r;
        float val = acc[tm][tn][r] + bv;
        float other = __shfl_xor(val, 1);       // even/odd column partner
        if (mode == 0) {
          dst[(size_t)row * EMB + col] = f2bf(val);
        } else {
          float outv = val;
          int t = row & (SEQ_T - 1);
          if (mode == 1) {
            float sn, cs;
            __sincosf((float)t * inv_freq, &sn, &cs);
            outv = ((idx & 1) == 0) ? (val * cs - other * sn)
                                    : (other * sn + val * cs);
          }
          int bb = row >> 11;                   // / SEQ_T
          dst[(((size_t)(bb * NHEAD + h) * SEQ_T) + t) * HDIM + d] = f2bf(outv);
        }
      }
    }
  }
}

// ---------------------------------------------------------------------------
// Flash attention, causal. One block = (b,h, 64-row q tile). 4 waves, 16 rows
// each. S and PV via v_wmma_f32_16x16x32_bf16, online softmax in f32.
// K tile staged by TDM (tensor_load_to_lds) when available.
// ---------------------------------------------------------------------------
__global__ __launch_bounds__(128, 2)
void flash_attn(const unsigned short* __restrict__ q,
                const unsigned short* __restrict__ k,
                const unsigned short* __restrict__ v,
                unsigned short* __restrict__ y)
{
  __shared__ unsigned short Ks[64 * LDK];   // [kv][d]
  __shared__ unsigned short Vs[64 * LDK];   // [d][kv] (transposed)
  __shared__ unsigned short Ps[64 * LDK];   // per-wave P slabs [qrow][kv]

  const int tid  = threadIdx.x;
  const int lane = tid & 31, wid = tid >> 5;
  const int half = lane >> 4, idx = lane & 15;

  const int qtile = blockIdx.x;
  const int bh = blockIdx.y;
  const int q0 = qtile * 64;

  const unsigned short* qp = q + (size_t)bh * SEQ_T * HDIM;
  const unsigned short* kp = k + (size_t)bh * SEQ_T * HDIM;
  const unsigned short* vp = v + (size_t)bh * SEQ_T * HDIM;

  // Q fragments for this wave (rows q0+wid*16 .. +15), D split into 2x K32
  Frag qa[2];
  {
    const unsigned short* qr = qp + (size_t)(q0 + wid * 16 + idx) * HDIM;
    #pragma unroll
    for (int d0 = 0; d0 < 2; ++d0)
      #pragma unroll
      for (int j = 0; j < 8; ++j) {
        int ko = d0 * 32 + (half << 4) + (j << 1);
        qa[d0].u[j] = *(const unsigned int*)(qr + ko);
      }
  }

  float mrow[8], lrow[8];
  f32x8 o[4];
  #pragma unroll
  for (int r = 0; r < 8; ++r) { mrow[r] = -1e30f; lrow[r] = 0.f; }
  #pragma unroll
  for (int tn = 0; tn < 4; ++tn)
    #pragma unroll
    for (int i = 0; i < 8; ++i) o[tn][i] = 0.f;

  const int rowbase = q0 + wid * 16 + half * 8;

  for (int kvt = 0; kvt <= qtile; ++kvt) {
    const int kv0 = kvt * 64;
    // ---- stage V (register transpose, all threads) ----
    {
      int r = tid >> 3, c = (tid & 7) * 8;
      #pragma unroll
      for (int i = 0; i < 4; ++i) {
        int row = r + 16 * i;
        unsigned short tmp[8];
        *(uint4*)tmp = *(const uint4*)(vp + (size_t)(kv0 + row) * HDIM + c);
        #pragma unroll
        for (int j = 0; j < 8; ++j) Vs[(c + j) * LDK + row] = tmp[j];
      }
      if (kvt < qtile) {  // hint next tile into cache (global_prefetch_b8)
        __builtin_prefetch(kp + (size_t)(kv0 + 64 + r) * HDIM, 0, 0);
        __builtin_prefetch(vp + (size_t)(kv0 + 64 + r) * HDIM, 0, 0);
      }
    }
    // ---- stage K: TDM tile DMA (wave 0) or async-b128 fallback ----
#if HAVE_TDM
    if (wid == 0) {
      tdm_load_tile64(&Ks[0], kp + (size_t)kv0 * HDIM);
      tdm_wait();
    }
#else
    {
      int r = tid >> 3, c = (tid & 7) * 8;
      #pragma unroll
      for (int i = 0; i < 4; ++i) {
        int row = r + 16 * i;
        async_copy16(&Ks[row * LDK + c],
                     kp + (size_t)(kv0 + row) * HDIM + c);
      }
      async_wait();
    }
#endif
    __syncthreads();

    // ---- S = Q K^T ----
    f32x8 s[4];
    #pragma unroll
    for (int tn = 0; tn < 4; ++tn)
      #pragma unroll
      for (int i = 0; i < 8; ++i) s[tn][i] = 0.f;
    #pragma unroll
    for (int d0 = 0; d0 < 2; ++d0)
      #pragma unroll
      for (int tn = 0; tn < 4; ++tn) {
        Frag kb;
        const unsigned short* p = &Ks[(tn * 16 + idx) * LDK + d0 * 32];
        #pragma unroll
        for (int j = 0; j < 8; ++j)
          kb.u[j] = *(const unsigned int*)(p + (half << 4) + (j << 1));
        s[tn] = __builtin_amdgcn_wmma_f32_16x16x32_bf16(
            false, qa[d0].v, false, kb.v, (short)0, s[tn], false, false);
      }

    // ---- scale + causal mask ----
    #pragma unroll
    for (int tn = 0; tn < 4; ++tn) {
      const int col = kv0 + tn * 16 + idx;
      #pragma unroll
      for (int r = 0; r < 8; ++r) {
        float sv = s[tn][r] * 0.125f;
        s[tn][r] = (col > rowbase + r) ? -10000.f : sv;
      }
    }

    // ---- online softmax (xor 1/2/4/8 stays within each 16-lane half) ----
    #pragma unroll
    for (int r = 0; r < 8; ++r) {
      float mx = fmaxf(fmaxf(s[0][r], s[1][r]), fmaxf(s[2][r], s[3][r]));
      mx = fmaxf(mx, __shfl_xor(mx, 1));
      mx = fmaxf(mx, __shfl_xor(mx, 2));
      mx = fmaxf(mx, __shfl_xor(mx, 4));
      mx = fmaxf(mx, __shfl_xor(mx, 8));
      float mnew  = fmaxf(mrow[r], mx);
      float alpha = __expf(mrow[r] - mnew);
      mrow[r] = mnew;
      float sum = 0.f;
      #pragma unroll
      for (int tn = 0; tn < 4; ++tn) {
        float pv = __expf(s[tn][r] - mnew);
        s[tn][r] = pv;
        sum += pv;
      }
      sum += __shfl_xor(sum, 1);
      sum += __shfl_xor(sum, 2);
      sum += __shfl_xor(sum, 4);
      sum += __shfl_xor(sum, 8);
      lrow[r] = lrow[r] * alpha + sum;
      #pragma unroll
      for (int tn = 0; tn < 4; ++tn) o[tn][r] *= alpha;
    }

    // ---- P -> LDS (per-wave slab), then O += P V ----
    unsigned short* Pw = &Ps[wid * 16 * LDK];
    #pragma unroll
    for (int tn = 0; tn < 4; ++tn)
      #pragma unroll
      for (int r = 0; r < 8; ++r)
        Pw[(half * 8 + r) * LDK + tn * 16 + idx] = f2bf(s[tn][r]);

    #pragma unroll
    for (int k2 = 0; k2 < 2; ++k2) {
      Frag pa;
      const unsigned short* pp = Pw + idx * LDK + k2 * 32;
      #pragma unroll
      for (int j = 0; j < 8; ++j) {
        int ko = (half << 4) + (j << 1);
        pa.u[j] = *(const unsigned int*)(pp + ko);
      }
      #pragma unroll
      for (int tn = 0; tn < 4; ++tn) {
        Frag vb;
        const unsigned short* vv = &Vs[(tn * 16 + idx) * LDK + k2 * 32];
        #pragma unroll
        for (int j = 0; j < 8; ++j)
          vb.u[j] = *(const unsigned int*)(vv + (half << 4) + (j << 1));
        o[tn] = __builtin_amdgcn_wmma_f32_16x16x32_bf16(
            false, pa.v, false, vb.v, (short)0, o[tn], false, false);
      }
    }
    __syncthreads();
  }

  // ---- normalize & store y[B,T,C] ----
  const int bb = bh >> 4;        // / NHEAD
  const int h  = bh & 15;
  #pragma unroll
  for (int r = 0; r < 8; ++r) {
    const float rcp = 1.f / lrow[r];
    const int t = rowbase + r;
    unsigned short* yr = y + ((size_t)(bb * SEQ_T + t)) * EMB + h * HDIM;
    #pragma unroll
    for (int tn = 0; tn < 4; ++tn)
      yr[tn * 16 + idx] = f2bf(o[tn][r] * rcp);
  }
}

// ---------------------------------------------------------------------------
extern "C" void kernel_launch(void* const* d_in, const int* in_sizes, int n_in,
                              void* d_out, int out_size, void* d_ws, size_t ws_size,
                              hipStream_t stream) {
  (void)in_sizes; (void)n_in; (void)out_size; (void)ws_size;
  const float*          x  = (const float*)d_in[0];
  const unsigned short* Wq = (const unsigned short*)d_in[1];
  const unsigned short* bq = (const unsigned short*)d_in[2];
  const unsigned short* Wk = (const unsigned short*)d_in[3];
  const unsigned short* bk = (const unsigned short*)d_in[4];
  const unsigned short* Wv = (const unsigned short*)d_in[5];
  const unsigned short* bv = (const unsigned short*)d_in[6];
  const unsigned short* Wo = (const unsigned short*)d_in[7];
  const unsigned short* bo = (const unsigned short*)d_in[8];
  unsigned short* out = (unsigned short*)d_out;

  const size_t qkvElems = (size_t)BATCH * NHEAD * SEQ_T * HDIM; // 8M elems
  unsigned short* xb = (unsigned short*)d_ws;          // x as bf16 (8M elems)
  unsigned short* qb = xb + qkvElems;
  unsigned short* kb = qb + qkvElems;
  unsigned short* vb = kb + qkvElems;
  unsigned short* yb = vb + qkvElems;   // total ws: 80 MB

  // one-shot f32 -> bf16 (8M elems, 4 per thread)
  cvt_f32_bf16<<<(MTOT * EMB) / 1024, 256, 0, stream>>>(x, xb);

  dim3 gg(EMB / BN, MTOT / BM);         // (16, 64)
  gemm_bf16_wmma<<<gg, 256, 0, stream>>>(xb, Wq, bq, qb, 1);
  gemm_bf16_wmma<<<gg, 256, 0, stream>>>(xb, Wk, bk, kb, 1);
  gemm_bf16_wmma<<<gg, 256, 0, stream>>>(xb, Wv, bv, vb, 2);

  dim3 ga(SEQ_T / 64, BATCH * NHEAD);   // (32, 64)
  flash_attn<<<ga, 128, 0, stream>>>(qb, kb, vb, yb);

  gemm_bf16_wmma<<<gg, 256, 0, stream>>>(yb, Wo, bo, out, 0);
}
